// Attention_23493471109335
// MI455X (gfx1250) — matrix operand order repaired and checked by reading, not compiled, not run on hardware
//
#include <hip/hip_runtime.h>
#include <math.h>

// ---------------------------------------------------------------------------
// Types
// ---------------------------------------------------------------------------
typedef __attribute__((ext_vector_type(16))) __bf16 v16bf;
typedef __attribute__((ext_vector_type(8)))  float  v8f;
typedef int v4i __attribute__((vector_size(16)));   // matches builtin's V4i

#if defined(__HIP_DEVICE_COMPILE__) && \
    __has_builtin(__builtin_amdgcn_global_load_async_to_lds_b128)
#define HAVE_ASYNC_LDS 1
#else
#define HAVE_ASYNC_LDS 0
#endif

__device__ inline v8f zero8() {
    v8f z;
    for (int e = 0; e < 8; ++e) z[e] = 0.0f;
    return z;
}

__device__ inline v8f wmma_bf16(v16bf a, v16bf b, v8f c) {
    // (neg_a, A, neg_b, B, c_mod, C, reuse_a, reuse_b)
    return __builtin_amdgcn_wmma_f32_16x16x32_bf16(false, a, false, b,
                                                   (short)0, c, false, false);
}

// 16-byte global -> LDS copy. Async path uses GLOBAL_LOAD_ASYNC_TO_LDS_B128
// (no VGPR data movement, tracked by ASYNCcnt); fallback is a sync copy.
__device__ inline void copy16_g2l(const __bf16* g, __bf16* l) {
#if HAVE_ASYNC_LDS
    __builtin_amdgcn_global_load_async_to_lds_b128(
        (__attribute__((address_space(1))) v4i*)(void*)g,
        (__attribute__((address_space(3))) v4i*)(void*)l, 0, 0);
#else
    *(uint4*)l = *(const uint4*)g;
#endif
}

__device__ inline void wait_g2l() {
#if HAVE_ASYNC_LDS
#if __has_builtin(__builtin_amdgcn_s_wait_asynccnt)
    __builtin_amdgcn_s_wait_asynccnt(0);
#else
    asm volatile("s_wait_asynccnt 0x0" ::: "memory");
#endif
#endif
}

// A-fragment (16x32 bf16). ISA 7.12.2: lane&15 = row M; lanes 0-15 hold
// K={0..7,16..23}, lanes 16-31 hold K={8..15,24..31}. base points at
// (row 0, k 0) of the 16x32 chunk; stride in elements.
__device__ inline v16bf load_frag_A(const __bf16* base, int stride, int lane) {
    union { uint4 u[2]; v16bf v; } f;
    const __bf16* p = base + (size_t)(lane & 15) * stride;
    int kg = (lane >> 4) * 8;
    f.u[0] = *(const uint4*)(p + kg);
    f.u[1] = *(const uint4*)(p + 16 + kg);
    return f.v;
}

// B-fragment (32x16 bf16), column-major source: memory row n holds the 32
// K-values of column n. lane&15 = column N; lanes 0-15 hold
// K={0..7,16..23}, lanes 16-31 hold K={8..15,24..31} (same as A).
__device__ inline v16bf load_frag_B(const __bf16* base, int stride, int lane) {
    union { uint4 u[2]; v16bf v; } f;
    const __bf16* p = base + (size_t)(lane & 15) * stride + (lane >> 4) * 8;
    f.u[0] = *(const uint4*)(p);
    f.u[1] = *(const uint4*)(p + 16);
    return f.v;
}

// ---------------------------------------------------------------------------
// fp32 -> bf16 conversion (4 elems / thread)
// ---------------------------------------------------------------------------
__global__ void cvt_f32_bf16(const float* __restrict__ in,
                             __bf16* __restrict__ out, int n) {
    int i = (blockIdx.x * blockDim.x + threadIdx.x) * 4;
    if (i + 3 < n) {
        float4 f = *(const float4*)(in + i);
        __align__(8) __bf16 t[4] = { (__bf16)f.x, (__bf16)f.y,
                                     (__bf16)f.z, (__bf16)f.w };
        *(uint2*)(out + i) = *(const uint2*)t;
    }
}

// ---------------------------------------------------------------------------
// GEMM: Y[M,N] = X[M,K] * W[N,K]^T   (both K-major), bf16 in, fp32 acc.
// Block tile 128x128, 8 waves (wave32), wave tile 32(M) x 64(N), K step 32.
// ---------------------------------------------------------------------------
#define G_LDSTR 40   // 32 + 8 pad (bf16 elems); 80B rows keep 16B alignment

template <typename OutT>
__global__ void __launch_bounds__(256)
gemm_xwT(const __bf16* __restrict__ X, const __bf16* __restrict__ W,
         OutT* __restrict__ Y, int M, int N, int K) {
    __shared__ __bf16 As[128 * G_LDSTR];
    __shared__ __bf16 Bs[128 * G_LDSTR];

    int tid  = threadIdx.x;
    int lane = tid & 31;
    int wave = tid >> 5;          // 0..7
    int wm   = wave & 3;          // 4 waves along M
    int wn   = wave >> 2;         // 2 waves along N
    int m0   = blockIdx.y * 128;
    int n0   = blockIdx.x * 128;
    int wrow = wm * 32;           // wave row offset inside block tile
    int wcol = wn * 64;           // wave col offset inside block tile

    v8f acc[2][4];
    for (int i = 0; i < 2; ++i)
        for (int j = 0; j < 4; ++j) acc[i][j] = zero8();

    for (int k0 = 0; k0 < K; k0 += 32) {
        __syncthreads();
        // stage 128x32 of X and 128x32 of W (rows of W = output columns)
        for (int t = 0; t < 2; ++t) {
            int idx = tid + t * 256;       // 0..511 uint4 slots
            int row = idx >> 2;            // 4 uint4 per 32-elem row
            int kc  = (idx & 3) * 8;
            copy16_g2l(&X[(size_t)(m0 + row) * K + k0 + kc],
                       &As[row * G_LDSTR + kc]);
            copy16_g2l(&W[(size_t)(n0 + row) * K + k0 + kc],
                       &Bs[row * G_LDSTR + kc]);
        }
        // prefetch next K-slice while this one lands
        if (k0 + 32 < K) {
            int pr = tid & 127;
            const __bf16* pp = (tid < 128)
                ? &X[(size_t)(m0 + pr) * K + k0 + 32]
                : &W[(size_t)(n0 + pr) * K + k0 + 32];
            __builtin_prefetch(pp, 0, 1);
        }
        wait_g2l();
        __syncthreads();

        v16bf a[2], b[4];
        for (int i = 0; i < 2; ++i)
            a[i] = load_frag_A(&As[(wrow + i * 16) * G_LDSTR], G_LDSTR, lane);
        for (int j = 0; j < 4; ++j)
            b[j] = load_frag_B(&Bs[(wcol + j * 16) * G_LDSTR], G_LDSTR, lane);

        for (int i = 0; i < 2; ++i)
            for (int j = 0; j < 4; ++j)
                acc[i][j] = wmma_bf16(a[i], b[j], acc[i][j]);
    }

    // D layout: VGPR r -> row (r + 8*(lane>>4)), col (lane&15)
    int lrow = 8 * (lane >> 4);
    int lcol = lane & 15;
    for (int i = 0; i < 2; ++i)
        for (int j = 0; j < 4; ++j) {
            int rbase = m0 + wrow + i * 16 + lrow;
            int c     = n0 + wcol + j * 16 + lcol;
            for (int r = 0; r < 8; ++r)
                Y[(size_t)(rbase + r) * N + c] = (OutT)acc[i][j][r];
        }
}

// ---------------------------------------------------------------------------
// RoPE + head transpose: [B,S,H*HD] -> [B*H, S, HD]   (1 complex pair/thread)
// ---------------------------------------------------------------------------
__global__ void rope_transpose(const __bf16* __restrict__ in,
                               __bf16* __restrict__ out,
                               const float* __restrict__ fcos,
                               const float* __restrict__ fsin, int do_rope) {
    int idx = blockIdx.x * blockDim.x + threadIdx.x;  // B*S*H*64 threads
    int i = idx & 63;             // pair index (HD/2 = 64)
    int t = idx >> 6;
    int h = t & 15;               // H = 16
    t >>= 4;
    int s = t & 2047;             // S = 2048
    int b = t >> 11;              // B = 2

    size_t src = ((size_t)(b * 2048 + s) * 2048) + h * 128 + 2 * i;
    float re = (float)in[src];
    float im = (float)in[src + 1];
    if (do_rope) {
        float c  = fcos[s * 64 + i];
        float sn = fsin[s * 64 + i];
        float nr = re * c - im * sn;
        float ni = re * sn + im * c;
        re = nr; im = ni;
    }
    size_t dst = ((size_t)(b * 16 + h) * 2048 + s) * 128 + 2 * i;
    out[dst]     = (__bf16)re;
    out[dst + 1] = (__bf16)im;
}

// ---------------------------------------------------------------------------
// Flash-style causal attention.
// Q,K,V: [B*H, S, HD=128] bf16.  Out: [B, S, H*HD] bf16 (natural layout).
// One block = one 128-row q-tile of one (b,h); 8 waves, 16 q-rows each.
// ---------------------------------------------------------------------------
#define A_STR 136   // 128 + 8 pad (bf16); 272B rows keep 16B alignment

__global__ void __launch_bounds__(256)
attention_kernel(const __bf16* __restrict__ Q, const __bf16* __restrict__ K,
                 const __bf16* __restrict__ V, __bf16* __restrict__ O) {
    __shared__ __bf16 Ks[128 * A_STR];   // [k_row][d]
    __shared__ __bf16 Vs[128 * A_STR];   // transposed: [d][k_row]
    __shared__ __bf16 Ps[128 * A_STR];   // probs, per-wave 16-row slices

    const int S = 2048;
    int tid  = threadIdx.x;
    int lane = tid & 31;
    int wave = tid >> 5;
    int qt   = blockIdx.x;               // q tile index (16)
    int bh   = blockIdx.y;               // b*16 + h    (32)
    int b    = bh >> 4;
    int h    = bh & 15;

    const __bf16* Qb = Q + (size_t)bh * S * 128;
    const __bf16* Kb = K + (size_t)bh * S * 128;
    const __bf16* Vb = V + (size_t)bh * S * 128;

    int q0   = qt * 128;
    int qrow = q0 + wave * 16;           // this wave's 16 q rows
    int lrow = 8 * (lane >> 4);
    int lcol = lane & 15;

    // Q fragments in registers: 4 chunks of d (32 each)
    v16bf qf[4];
    for (int dc = 0; dc < 4; ++dc)
        qf[dc] = load_frag_A(Qb + (size_t)qrow * 128 + dc * 32, 128, lane);

    v8f   o_acc[8];
    float m_run[8], l_run[8];
    for (int d = 0; d < 8; ++d) o_acc[d] = zero8();
    for (int r = 0; r < 8; ++r) { m_run[r] = -1e30f; l_run[r] = 0.0f; }

    const float scale = 0.08838834764831845f;   // 1/sqrt(128)

    int ktiles = qt + 1;                 // causal
    for (int kt = 0; kt < ktiles; ++kt) {
        int kbase = kt * 128;
        __syncthreads();                 // prior-iter LDS readers done
        // stage K (async, row-major) and V (transposed): 2048 uint4, 8/thread
        for (int t = 0; t < 8; ++t) {
            int idx = tid + t * 256;
            int row = idx >> 4;
            int d8  = (idx & 15) * 8;
            copy16_g2l(&Kb[(size_t)(kbase + row) * 128 + d8],
                       &Ks[row * A_STR + d8]);
            uint4 vv = *(const uint4*)&Vb[(size_t)(kbase + row) * 128 + d8];
            const __bf16* ve = (const __bf16*)&vv;
            for (int j = 0; j < 8; ++j)
                Vs[(d8 + j) * A_STR + row] = ve[j];
        }
        wait_g2l();
        __syncthreads();

        // scores: 16 x 128 per wave, 8 col-subtiles x 4 d-chunks of WMMA
        v8f sc[8];
        for (int nt = 0; nt < 8; ++nt) {
            v8f s = zero8();
            for (int dc = 0; dc < 4; ++dc) {
                v16bf bk = load_frag_B(&Ks[(nt * 16) * A_STR + dc * 32],
                                       A_STR, lane);
                s = wmma_bf16(qf[dc], bk, s);
            }
            sc[nt] = s;
        }

        // scale + causal mask + online softmax (row lives across 16 lanes)
        for (int r = 0; r < 8; ++r) {
            int grow = qrow + lrow + r;
            float mx = -1e30f;
            for (int nt = 0; nt < 8; ++nt) {
                float v = sc[nt][r] * scale;
                int gcol = kbase + nt * 16 + lcol;
                if (gcol > grow) v = -1e9f;
                sc[nt][r] = v;
                mx = fmaxf(mx, v);
            }
            for (int off = 1; off < 16; off <<= 1)
                mx = fmaxf(mx, __shfl_xor(mx, off, 32));
            float mnew = fmaxf(m_run[r], mx);
            float corr = __expf(m_run[r] - mnew);
            float rs = 0.0f;
            for (int nt = 0; nt < 8; ++nt) {
                float p = __expf(sc[nt][r] - mnew);
                sc[nt][r] = p;
                rs += p;
            }
            for (int off = 1; off < 16; off <<= 1)
                rs += __shfl_xor(rs, off, 32);
            l_run[r] = l_run[r] * corr + rs;
            m_run[r] = mnew;
            for (int d = 0; d < 8; ++d) o_acc[d][r] *= corr;
        }

        // P -> LDS (wave-private 16-row slice), reshaped for A-fragments
        for (int nt = 0; nt < 8; ++nt)
            for (int r = 0; r < 8; ++r)
                Ps[(wave * 16 + lrow + r) * A_STR + nt * 16 + lcol] =
                    (__bf16)sc[nt][r];
        __syncthreads();

        // O += P(16x128) * V(128x16 per d-subtile)
        for (int d = 0; d < 8; ++d) {
            v8f oa = o_acc[d];
            for (int kc = 0; kc < 4; ++kc) {
                v16bf pa = load_frag_A(&Ps[(wave * 16) * A_STR + kc * 32],
                                       A_STR, lane);
                v16bf vb = load_frag_B(&Vs[(d * 16) * A_STR + kc * 32],
                                       A_STR, lane);
                oa = wmma_bf16(pa, vb, oa);
            }
            o_acc[d] = oa;
        }
    }

    // normalize and write back to natural [B,S,D] layout
    for (int d = 0; d < 8; ++d)
        for (int r = 0; r < 8; ++r) {
            int row = qrow + lrow + r;
            float v = o_acc[d][r] / l_run[r];
            O[((size_t)b * S + row) * 2048 + h * 128 + d * 16 + lcol] =
                (__bf16)v;
        }
}

// ---------------------------------------------------------------------------
// Host launch
// ---------------------------------------------------------------------------
extern "C" void kernel_launch(void* const* d_in, const int* in_sizes, int n_in,
                              void* d_out, int out_size, void* d_ws,
                              size_t ws_size, hipStream_t stream) {
    (void)in_sizes; (void)n_in; (void)out_size; (void)ws_size;
    const float* x  = (const float*)d_in[0];
    const float* wq = (const float*)d_in[1];
    const float* wk = (const float*)d_in[2];
    const float* wv = (const float*)d_in[3];
    const float* wo = (const float*)d_in[4];
    const float* fc = (const float*)d_in[5];
    const float* fs = (const float*)d_in[6];
    // d_in[7] = mask, implemented arithmetically (causal) in-kernel
    float* out = (float*)d_out;

    const size_t XB = (size_t)2 * 2048 * 2048;   // B*S*D elements
    const size_t WE = (size_t)2048 * 2048;

    __bf16* p   = (__bf16*)d_ws;
    __bf16* xb  = p; p += XB;
    __bf16* wqb = p; p += WE;
    __bf16* wkb = p; p += WE;
    __bf16* wvb = p; p += WE;
    __bf16* wob = p; p += WE;
    __bf16* qn  = p; p += XB;
    __bf16* kn  = p; p += XB;
    __bf16* vn  = p; p += XB;
    __bf16* kt_ = p; p += XB;
    __bf16* vt_ = p; p += XB;
    __bf16* qt_ = xb;   // xb dead after the three projections
    __bf16* ao  = qn;   // qn dead after rope(q)

    // 1) convert fp32 -> bf16
    cvt_f32_bf16<<<(unsigned)(XB / 1024), 256, 0, stream>>>(x, xb, (int)XB);
    cvt_f32_bf16<<<(unsigned)(WE / 1024), 256, 0, stream>>>(wq, wqb, (int)WE);
    cvt_f32_bf16<<<(unsigned)(WE / 1024), 256, 0, stream>>>(wk, wkb, (int)WE);
    cvt_f32_bf16<<<(unsigned)(WE / 1024), 256, 0, stream>>>(wv, wvb, (int)WE);
    cvt_f32_bf16<<<(unsigned)(WE / 1024), 256, 0, stream>>>(wo, wob, (int)WE);

    // 2) Q/K/V projections: [4096,2048] @ [2048,2048]^T
    dim3 gg(2048 / 128, 4096 / 128);
    gemm_xwT<__bf16><<<gg, 256, 0, stream>>>(xb, wqb, qn, 4096, 2048, 2048);
    gemm_xwT<__bf16><<<gg, 256, 0, stream>>>(xb, wkb, kn, 4096, 2048, 2048);
    gemm_xwT<__bf16><<<gg, 256, 0, stream>>>(xb, wvb, vn, 4096, 2048, 2048);

    // 3) RoPE + transpose to [B*H, S, HD]
    unsigned rgrid = (unsigned)((size_t)2 * 2048 * 16 * 64 / 256);
    rope_transpose<<<rgrid, 256, 0, stream>>>(qn, qt_, fc, fs, 1);
    rope_transpose<<<rgrid, 256, 0, stream>>>(kn, kt_, fc, fs, 1);
    rope_transpose<<<rgrid, 256, 0, stream>>>(vn, vt_, fc, fs, 0);

    // 4) causal flash attention -> natural [B,S,D] bf16
    dim3 ag(2048 / 128, 2 * 16);
    attention_kernel<<<ag, 256, 0, stream>>>(qt_, kt_, vt_, ao);

    // 5) output projection (fp32 out)
    gemm_xwT<float><<<gg, 256, 0, stream>>>(ao, wob, out, 4096, 2048, 2048);
}
